// TemporalGATModel_54494545051654
// MI455X (gfx1250) — matrix operand order repaired and checked by reading, not compiled, not run on hardware
//
#include <hip/hip_runtime.h>
#include <hip/hip_bf16.h>

typedef __attribute__((ext_vector_type(16))) _Float16 v16h;
typedef __attribute__((ext_vector_type(8)))  _Float16 v8h;
typedef __attribute__((ext_vector_type(8)))  float    v8f;

constexpr int       NN   = 50000;
constexpr int       TT   = 16;
constexpr int       DD   = 4;
constexpr long long EE   = 800000;
constexpr int       HH   = 4;
constexpr int       HID  = 32;   // per-head GAT dim and model dim
constexpr int       THID = 64;   // temporal attention total dim
constexpr int       HD   = 16;   // temporal head dim
constexpr float     NEG_SLOPE = 0.2f;

// ---------- order-preserving float <-> uint for atomic max ----------
static __device__ __forceinline__ unsigned fkey(float f) {
  unsigned u = __float_as_uint(f);
  return (u & 0x80000000u) ? ~u : (u | 0x80000000u);
}
static __device__ __forceinline__ float fdec(unsigned k) {
  unsigned u = (k & 0x80000000u) ? (k ^ 0x80000000u) : ~k;
  return __uint_as_float(u);
}

// ---------- WMMA tile loaders ----------
// A: 16x32 (MxK) f16 layout: lanes 0-15 row M=lane, K {0..7,16..23};
// lanes 16-31 same rows, K {8..15,24..31}.  All offsets 16B aligned.
static __device__ __forceinline__ v16h load_A_tile(const float* __restrict__ A,
                                                   long long lda, int lane) {
  int half = lane >> 4;
  int row  = lane & 15;
  const float* p = A + (long long)row * lda + half * 8;
  float4 f0 = *(const float4*)(p);
  float4 f1 = *(const float4*)(p + 4);
  float4 f2 = *(const float4*)(p + 16);
  float4 f3 = *(const float4*)(p + 20);
  v16h a;
  a[0]  = (_Float16)f0.x;  a[1]  = (_Float16)f0.y;
  a[2]  = (_Float16)f0.z;  a[3]  = (_Float16)f0.w;
  a[4]  = (_Float16)f1.x;  a[5]  = (_Float16)f1.y;
  a[6]  = (_Float16)f1.z;  a[7]  = (_Float16)f1.w;
  a[8]  = (_Float16)f2.x;  a[9]  = (_Float16)f2.y;
  a[10] = (_Float16)f2.z;  a[11] = (_Float16)f2.w;
  a[12] = (_Float16)f3.x;  a[13] = (_Float16)f3.y;
  a[14] = (_Float16)f3.z;  a[15] = (_Float16)f3.w;
  return a;
}
static __device__ __forceinline__ v16h load_A_tile(const _Float16* __restrict__ A,
                                                   long long lda, int lane) {
  int half = lane >> 4;
  int row  = lane & 15;
  const _Float16* p = A + (long long)row * lda + half * 8;
  v8h lo = *(const v8h*)(p);
  v8h hi = *(const v8h*)(p + 16);
  v16h a;
#pragma unroll
  for (int i = 0; i < 8; ++i) { a[i] = lo[i]; a[8 + i] = hi[i]; }
  return a;
}

// B: 32x16 (KxN) f16 layout: lanes 0-15 hold K {0..7,16..23} (col N=lane), lanes 16-31 K {8..15,24..31}.
// Loads are column-strided per lane but fully coalesced across lanes; B is tiny (L1).
static __device__ __forceinline__ v16h load_B_tile(const float* __restrict__ B,
                                                   int ldb, int lane) {
  int half = lane >> 4;
  int col  = lane & 15;
  const float* p = B + (long long)(half * 8) * ldb + col;
  v16h b;
#pragma unroll
  for (int v = 0; v < 16; ++v) b[v] = (_Float16)p[(long long)(v < 8 ? v : v + 8) * ldb];
  return b;
}

// C/D 16x16 f32 layout: VGPR r -> row M = r + 8*(lane>=16), col N = lane&15.
static __device__ __forceinline__ void store_tile_f32(float* __restrict__ C, int ldc,
                                                      long long rowTile, int colTile,
                                                      int lane, const v8f& c, float bv) {
  int half = lane >> 4;
  int col  = lane & 15;
#pragma unroll
  for (int r = 0; r < 8; ++r)
    C[(rowTile + r + 8 * half) * (long long)ldc + colTile + col] = c[r] + bv;
}
static __device__ __forceinline__ void store_tile_f16(_Float16* __restrict__ C, int ldc,
                                                      long long rowTile, int colTile,
                                                      int lane, const v8f& c, float bv) {
  int half = lane >> 4;
  int col  = lane & 15;
#pragma unroll
  for (int r = 0; r < 8; ++r)
    C[(rowTile + r + 8 * half) * (long long)ldc + colTile + col] = (_Float16)(c[r] + bv);
}

// ---------- generic WMMA GEMM:  C(M,Nout) = A(M,K) @ B(K,Nout) + bias ----------
template <typename TA, bool HALF_OUT>
__global__ void wmma_gemm(const TA* __restrict__ A, const float* __restrict__ B,
                          const float* __restrict__ bias, void* __restrict__ Cout,
                          long long M, int Nout, int K) {
  int lane = threadIdx.x & 31;
  int wave = threadIdx.x >> 5;
  long long rowTile = ((long long)blockIdx.x * 4 + wave) * 16;
  int colTile = blockIdx.y * 16;
  if (rowTile >= M) return;   // wave-uniform: EXEC stays all-ones for WMMA

  v8f c = {};
  for (int k0 = 0; k0 < K; k0 += 32) {
    v16h a = load_A_tile(A + rowTile * (long long)K + k0, K, lane);
    v16h b = load_B_tile(B + (long long)k0 * Nout + colTile, Nout, lane);
    c = __builtin_amdgcn_wmma_f32_16x16x32_f16(false, a, false, b,
                                               (short)0, c, false, false);
  }
  float bv = bias ? bias[colTile + (lane & 15)] : 0.0f;
  if (HALF_OUT) store_tile_f16((_Float16*)Cout, Nout, rowTile, colTile, lane, c, bv);
  else          store_tile_f32((float*)Cout,    Nout, rowTile, colTile, lane, c, bv);
}

// ---------- fused GAT xl/xr GEMM: A tile (x_pool) loaded once, 2 WMMAs ----------
__global__ void wmma_gat_lr(const float* __restrict__ A,
                            const float* __restrict__ Wl, const float* __restrict__ bl,
                            const float* __restrict__ Wr, const float* __restrict__ br,
                            float* __restrict__ xl, float* __restrict__ xr) {
  int lane = threadIdx.x & 31;
  int wave = threadIdx.x >> 5;
  long long rowTile = ((long long)blockIdx.x * 4 + wave) * 16;
  int colTile = blockIdx.y * 16;
  if (rowTile >= NN) return;
  constexpr int NO = HH * HID;  // 128
  v16h a  = load_A_tile(A + rowTile * (long long)HID, HID, lane);
  v16h b0 = load_B_tile(Wl + colTile, NO, lane);
  v16h b1 = load_B_tile(Wr + colTile, NO, lane);
  v8f c0 = {}, c1 = {};
  c0 = __builtin_amdgcn_wmma_f32_16x16x32_f16(false, a, false, b0, (short)0, c0, false, false);
  c1 = __builtin_amdgcn_wmma_f32_16x16x32_f16(false, a, false, b1, (short)0, c1, false, false);
  int col = lane & 15;
  store_tile_f32(xl, NO, rowTile, colTile, lane, c0, bl[colTile + col]);
  store_tile_f32(xr, NO, rowTile, colTile, lane, c1, br[colTile + col]);
}

// ---------- fused Q/K/V GEMM: A tile (x_seq) loaded once, 3 WMMAs ----------
__global__ void wmma_qkv(const float* __restrict__ A,
                         const float* __restrict__ Wq, const float* __restrict__ bq,
                         const float* __restrict__ Wk, const float* __restrict__ bk,
                         const float* __restrict__ Wv, const float* __restrict__ bv,
                         _Float16* __restrict__ qb, _Float16* __restrict__ kb,
                         _Float16* __restrict__ vb, long long M) {
  int lane = threadIdx.x & 31;
  int wave = threadIdx.x >> 5;
  long long rowTile = ((long long)blockIdx.x * 4 + wave) * 16;
  int colTile = blockIdx.y * 16;
  if (rowTile >= M) return;
  v16h a  = load_A_tile(A + rowTile * (long long)HID, HID, lane);
  v16h b0 = load_B_tile(Wq + colTile, THID, lane);
  v16h b1 = load_B_tile(Wk + colTile, THID, lane);
  v16h b2 = load_B_tile(Wv + colTile, THID, lane);
  v8f c0 = {}, c1 = {}, c2 = {};
  c0 = __builtin_amdgcn_wmma_f32_16x16x32_f16(false, a, false, b0, (short)0, c0, false, false);
  c1 = __builtin_amdgcn_wmma_f32_16x16x32_f16(false, a, false, b1, (short)0, c1, false, false);
  c2 = __builtin_amdgcn_wmma_f32_16x16x32_f16(false, a, false, b2, (short)0, c2, false, false);
  int col = lane & 15;
  store_tile_f16(qb, THID, rowTile, colTile, lane, c0, bq[colTile + col]);
  store_tile_f16(kb, THID, rowTile, colTile, lane, c1, bk[colTile + col]);
  store_tile_f16(vb, THID, rowTile, colTile, lane, c2, bv[colTile + col]);
}

// ---------- input projection: xs[t*N+n, c] = x[n,t,:] @ in_W + in_b ----------
__global__ void k_input_proj(const float* __restrict__ x, const float* __restrict__ W,
                             const float* __restrict__ b, float* __restrict__ xs) {
  int i = blockIdx.x * blockDim.x + threadIdx.x;   // over N*HID
  if (i >= NN * HID) return;
  int c = i & (HID - 1);
  int n = i >> 5;
  int t = blockIdx.y;
  float4 xr = *(const float4*)(x + ((long long)n * TT + t) * DD);
  float acc = b[c] + xr.x * W[c] + xr.y * W[HID + c] + xr.z * W[2 * HID + c]
            + xr.w * W[3 * HID + c];
  xs[((long long)t * NN + n) * HID + c] = acc;
}

__global__ void k_pool(const float* __restrict__ xs, float* __restrict__ xp) {
  int i = blockIdx.x * blockDim.x + threadIdx.x;   // over N*HID
  if (i >= NN * HID) return;
  int c = i & (HID - 1);
  int n = i >> 5;
  float s = 0.f;
#pragma unroll
  for (int t = 0; t < TT; ++t) s += xs[((long long)t * NN + n) * HID + c];
  xp[i] = s * (1.0f / TT);
}

__global__ void k_bcast_add(float* __restrict__ xs, const float* __restrict__ xp) {
  int i = blockIdx.x * blockDim.x + threadIdx.x;   // over N*HID
  if (i >= NN * HID) return;
  int c = i & (HID - 1);
  int n = i >> 5;
  int t = blockIdx.y;
  xs[((long long)t * NN + n) * HID + c] += xp[i];
}

__global__ void k_zero(float* __restrict__ p, long long n) {
  long long i = (long long)blockIdx.x * blockDim.x + threadIdx.x;
  if (i < n) p[i] = 0.0f;
}

// ---------- GATv2 edge kernels (float4-vectorized 32-float head slices) ----------
__global__ void k_gat_logits(const int* __restrict__ src, const int* __restrict__ dst,
                             const float* __restrict__ ew, const float* __restrict__ xl,
                             const float* __restrict__ xr, const float* __restrict__ We,
                             const float* __restrict__ att, float* __restrict__ elog,
                             unsigned* __restrict__ mxk) {
  long long tid = (long long)blockIdx.x * blockDim.x + threadIdx.x;
  if (tid >= EE * HH) return;
  long long e = tid >> 2;
  int h = (int)(tid & 3);
  int s = src[e], d = dst[e];
  float w = ew[e];
  const float4* pl = (const float4*)(xl + (long long)s * (HH * HID) + h * HID);
  const float4* pr = (const float4*)(xr + (long long)d * (HH * HID) + h * HID);
  const float4* pe = (const float4*)(We + h * HID);
  const float4* pa = (const float4*)(att + h * HID);
  float acc = 0.f;
#pragma unroll
  for (int j = 0; j < HID / 4; ++j) {
    float4 l = pl[j], r = pr[j], ev = pe[j], av = pa[j];
    float m0 = l.x + r.x + w * ev.x; m0 = (m0 > 0.f) ? m0 : NEG_SLOPE * m0;
    float m1 = l.y + r.y + w * ev.y; m1 = (m1 > 0.f) ? m1 : NEG_SLOPE * m1;
    float m2 = l.z + r.z + w * ev.z; m2 = (m2 > 0.f) ? m2 : NEG_SLOPE * m2;
    float m3 = l.w + r.w + w * ev.w; m3 = (m3 > 0.f) ? m3 : NEG_SLOPE * m3;
    acc += m0 * av.x + m1 * av.y + m2 * av.z + m3 * av.w;
  }
  elog[tid] = acc;
  atomicMax(&mxk[(long long)d * HH + h], fkey(acc));
}

__global__ void k_gat_exp(const int* __restrict__ dst, float* __restrict__ elog,
                          const unsigned* __restrict__ mxk, float* __restrict__ denom) {
  long long tid = (long long)blockIdx.x * blockDim.x + threadIdx.x;
  if (tid >= EE * HH) return;
  long long e = tid >> 2;
  int h = (int)(tid & 3);
  int d = dst[e];
  float mx = fdec(mxk[(long long)d * HH + h]);
  float a = expf(elog[tid] - mx);
  elog[tid] = a;
  atomicAdd(&denom[(long long)d * HH + h], a);
}

__global__ void k_gat_aggr(const int* __restrict__ src, const int* __restrict__ dst,
                           const float* __restrict__ elog, const float* __restrict__ denom,
                           const float* __restrict__ xl, float* __restrict__ outacc) {
  long long tid = (long long)blockIdx.x * blockDim.x + threadIdx.x;
  if (tid >= EE * HH) return;
  long long e = tid >> 2;
  int h = (int)(tid & 3);
  int s = src[e], d = dst[e];
  float al = elog[tid] / (denom[(long long)d * HH + h] + 1e-16f);
  const float4* pl = (const float4*)(xl + (long long)s * (HH * HID) + h * HID);
  float* po = outacc + (long long)d * (HH * HID) + h * HID;
#pragma unroll
  for (int j = 0; j < HID / 4; ++j) {
    float4 l = pl[j];
    atomicAdd(&po[4 * j + 0], al * l.x);
    atomicAdd(&po[4 * j + 1], al * l.y);
    atomicAdd(&po[4 * j + 2], al * l.z);
    atomicAdd(&po[4 * j + 3], al * l.w);
  }
}

__global__ void k_gat_finish(const float* __restrict__ outacc,
                             const float* __restrict__ bias, float* __restrict__ xp) {
  int i = blockIdx.x * blockDim.x + threadIdx.x;   // over N*HID
  if (i >= NN * HID) return;
  int c = i & (HID - 1);
  int n = i >> 5;
  float s = 0.f;
#pragma unroll
  for (int h = 0; h < HH; ++h) s += outacc[(long long)n * (HH * HID) + h * HID + c];
  float val = s * (1.0f / HH) + bias[c];
  xp[i] = fmaxf(val, 0.0f);
}

// ---------- per-(node, head, t) temporal attention; ctx overwrites Q ----------
__global__ void k_attn(const _Float16* q, const _Float16* __restrict__ k,
                       const _Float16* __restrict__ v, _Float16* ctx) {
  long long tid = (long long)blockIdx.x * blockDim.x + threadIdx.x;
  if (tid >= (long long)NN * HH * TT) return;
  int t = (int)(tid & 15);
  int h = (int)((tid >> 4) & 3);
  long long n = tid >> 6;

  const _Float16* qp = q + ((long long)t * NN + n) * THID + h * HD;
  v8h q0 = *(const v8h*)(qp);
  v8h q1 = *(const v8h*)(qp + 8);
  float qv[HD];
#pragma unroll
  for (int d = 0; d < 8; ++d) { qv[d] = (float)q0[d]; qv[8 + d] = (float)q1[d]; }

  float sc[TT];
  float mx = -1e30f;
#pragma unroll
  for (int s = 0; s < TT; ++s) {
    const _Float16* kp = k + ((long long)s * NN + n) * THID + h * HD;
    v8h k0 = *(const v8h*)(kp);
    v8h k1 = *(const v8h*)(kp + 8);
    float a = 0.f;
#pragma unroll
    for (int d = 0; d < 8; ++d) a += qv[d] * (float)k0[d] + qv[8 + d] * (float)k1[d];
    a *= 0.25f;   // HD^-0.5
    sc[s] = a;
    mx = fmaxf(mx, a);
  }
  float sum = 0.f;
#pragma unroll
  for (int s = 0; s < TT; ++s) { sc[s] = expf(sc[s] - mx); sum += sc[s]; }
  float inv = 1.0f / sum;

  float acc[HD];
#pragma unroll
  for (int d = 0; d < HD; ++d) acc[d] = 0.f;
#pragma unroll
  for (int s = 0; s < TT; ++s) {
    const _Float16* vp = v + ((long long)s * NN + n) * THID + h * HD;
    v8h v0 = *(const v8h*)(vp);
    v8h v1 = *(const v8h*)(vp + 8);
    float w = sc[s] * inv;
#pragma unroll
    for (int d = 0; d < 8; ++d) {
      acc[d]     += w * (float)v0[d];
      acc[8 + d] += w * (float)v1[d];
    }
  }
  v8h o0, o1;
#pragma unroll
  for (int d = 0; d < 8; ++d) { o0[d] = (_Float16)acc[d]; o1[d] = (_Float16)acc[8 + d]; }
  _Float16* cp = ctx + ((long long)t * NN + n) * THID + h * HD;
  *(v8h*)(cp)     = o0;
  *(v8h*)(cp + 8) = o1;
}

// ---------- forecast / risk heads: one wave per node, lane = hidden unit ----------
__global__ void k_heads(const float* __restrict__ tout,
                        const float* __restrict__ fW1, const float* __restrict__ fb1,
                        const float* __restrict__ fW2, const float* __restrict__ fb2,
                        const float* __restrict__ rW1, const float* __restrict__ rb1,
                        const float* __restrict__ rW2, const float* __restrict__ rb2,
                        float* __restrict__ out) {
  int wave = threadIdx.x >> 5;
  int lane = threadIdx.x & 31;
  long long n = (long long)blockIdx.x * 8 + wave;
  if (n >= NN) return;
  const float* last = tout + ((long long)(TT - 1) * NN + n) * HID;
  float hf = fb1[lane], hr = rb1[lane];
#pragma unroll
  for (int c = 0; c < HID; ++c) {
    float lv = last[c];
    hf += lv * fW1[c * 32 + lane];
    hr += lv * rW1[c * 32 + lane];
  }
  hf = fmaxf(hf, 0.f) * fW2[lane];
  hr = fmaxf(hr, 0.f) * rW2[lane];
#pragma unroll
  for (int off = 16; off > 0; off >>= 1) {
    hf += __shfl_xor(hf, off, 32);
    hr += __shfl_xor(hr, off, 32);
  }
  if (lane == 0) {
    out[n]      = fmaxf(hf + fb2[0], 0.0f);
    out[NN + n] = 1.0f / (1.0f + expf(-(hr + rb2[0])));
  }
}

extern "C" void kernel_launch(void* const* d_in, const int* in_sizes, int n_in,
                              void* d_out, int out_size, void* d_ws, size_t ws_size,
                              hipStream_t stream) {
  (void)in_sizes; (void)n_in; (void)out_size; (void)ws_size;

  const float* x   = (const float*)d_in[0];
  const int*   ei  = (const int*)d_in[1];
  const float* ew  = (const float*)d_in[2];
  const float* inW = (const float*)d_in[3];
  const float* inb = (const float*)d_in[4];
  const float* gWl[2]   = {(const float*)d_in[5],  (const float*)d_in[12]};
  const float* gbl[2]   = {(const float*)d_in[6],  (const float*)d_in[13]};
  const float* gWr[2]   = {(const float*)d_in[7],  (const float*)d_in[14]};
  const float* gbr[2]   = {(const float*)d_in[8],  (const float*)d_in[15]};
  const float* gWe[2]   = {(const float*)d_in[9],  (const float*)d_in[16]};
  const float* gatt[2]  = {(const float*)d_in[10], (const float*)d_in[17]};
  const float* gbias[2] = {(const float*)d_in[11], (const float*)d_in[18]};
  const float* Wq = (const float*)d_in[19]; const float* bq = (const float*)d_in[20];
  const float* Wk = (const float*)d_in[21]; const float* bk = (const float*)d_in[22];
  const float* Wv = (const float*)d_in[23]; const float* bv = (const float*)d_in[24];
  const float* Wo = (const float*)d_in[25]; const float* bo = (const float*)d_in[26];
  const float* fW1 = (const float*)d_in[27]; const float* fb1 = (const float*)d_in[28];
  const float* fW2 = (const float*)d_in[29]; const float* fb2 = (const float*)d_in[30];
  const float* rW1 = (const float*)d_in[31]; const float* rb1 = (const float*)d_in[32];
  const float* rW2 = (const float*)d_in[33]; const float* rb2 = (const float*)d_in[34];

  const int* srcIdx = ei;
  const int* dstIdx = ei + EE;

  // ---- workspace layout (floats) ----
  float* p = (float*)d_ws;
  float* xs     = p; p += (long long)TT * NN * HID;     // 25.6M  (t-major rows)
  float* xp     = p; p += (long long)NN * HID;          // 1.6M
  float* xl     = p; p += (long long)NN * HH * HID;     // 6.4M
  float* xrg    = p; p += (long long)NN * HH * HID;     // 6.4M
  float* elog   = p; p += EE * HH;                      // 3.2M
  float* zbase  = p;                                    // contiguous zeroed region:
  unsigned* mxk = (unsigned*)p; p += (long long)NN * HH;
  float* denom  = p; p += (long long)NN * HH;
  float* outacc = p; p += (long long)NN * HH * HID;
  long long zcount = (long long)NN * HH * 2 + (long long)NN * HH * HID;
  _Float16* qb = (_Float16*)p; p += (long long)TT * NN * THID / 2;
  _Float16* kb = (_Float16*)p; p += (long long)TT * NN * THID / 2;
  _Float16* vb = (_Float16*)p; p += (long long)TT * NN * THID / 2;

  float* tout = (float*)d_out + 2 * NN;  // temporal_out region of d_out

  dim3 b256(256);
  long long eh = EE * HH;

  // 1) x_seq = x @ in_W + in_b   (stored t-major: row = t*N+n)
  { dim3 g((NN * HID + 255) / 256, TT);
    k_input_proj<<<g, b256, 0, stream>>>(x, inW, inb, xs); }

  // 2) x_pool = mean_t x_seq
  { dim3 g((NN * HID + 255) / 256);
    k_pool<<<g, b256, 0, stream>>>(xs, xp); }

  // 3) two GATv2 layers (x_pool updated in place by k_gat_finish)
  const int rowTilesN = NN / 16;                         // 3125
  for (int L = 0; L < 2; ++L) {
    dim3 gg((rowTilesN + 3) / 4, (HH * HID) / 16);       // 782 x 8
    wmma_gat_lr<<<gg, dim3(128), 0, stream>>>(xp, gWl[L], gbl[L], gWr[L], gbr[L], xl, xrg);
    k_zero<<<dim3((unsigned)((zcount + 255) / 256)), b256, 0, stream>>>(zbase, zcount);
    k_gat_logits<<<dim3((unsigned)((eh + 255) / 256)), b256, 0, stream>>>(
        srcIdx, dstIdx, ew, xl, xrg, gWe[L], gatt[L], elog, mxk);
    k_gat_exp<<<dim3((unsigned)((eh + 255) / 256)), b256, 0, stream>>>(
        dstIdx, elog, mxk, denom);
    k_gat_aggr<<<dim3((unsigned)((eh + 255) / 256)), b256, 0, stream>>>(
        srcIdx, dstIdx, elog, denom, xl, outacc);
    k_gat_finish<<<dim3((NN * HID + 255) / 256), b256, 0, stream>>>(outacc, gbias[L], xp);
  }

  // 4) x_seq += x_pool (broadcast over t)
  { dim3 g((NN * HID + 255) / 256, TT);
    k_bcast_add<<<g, b256, 0, stream>>>(xs, xp); }

  // 5) fused Q/K/V projections (M = T*N = 800000, K=32, Nout=64), f16 outputs
  const long long Mtn = (long long)TT * NN;
  { dim3 gq((unsigned)(Mtn / 16 / 4), THID / 16);        // 12500 x 4
    wmma_qkv<<<gq, dim3(128), 0, stream>>>(xs, Wq, bq, Wk, bk, Wv, bv,
                                           qb, kb, vb, Mtn); }

  // 6) per-(node, head) 16x16 attention; ctx written over qb (safe in-place)
  { long long tot = (long long)NN * HH * TT;
    k_attn<<<dim3((unsigned)((tot + 255) / 256)), b256, 0, stream>>>(qb, kb, vb, qb); }

  // 7) temporal_out = ctx @ Wo + bo  -> directly into d_out (K=64, Nout=32)
  { dim3 go((unsigned)(Mtn / 16 / 4), HID / 16);         // 12500 x 2
    wmma_gemm<_Float16, false><<<go, dim3(128), 0, stream>>>(qb, Wo, bo, tout,
                                                             Mtn, HID, THID); }

  // 8) forecast / risk heads (one wave per node)
  { dim3 g((NN + 7) / 8);
    k_heads<<<g, b256, 0, stream>>>(tout, fW1, fb1, fW2, fb2,
                                    rW1, rb1, rW2, rb2, (float*)d_out); }
}
